// NER_Model_23819888623755
// MI455X (gfx1250) — matrix operand order repaired and checked by reading, not compiled, not run on hardware
//
#include <hip/hip_runtime.h>
#include <hip/hip_bf16.h>
#include <math.h>

typedef __attribute__((ext_vector_type(16))) _Float16 v16h;
typedef __attribute__((ext_vector_type(8)))  _Float16 v8h;
typedef __attribute__((ext_vector_type(2)))  _Float16 h2;
typedef __attribute__((ext_vector_type(8)))  float    v8f;

#define T_LEN 4096
#define HID   512
#define GATES 2048   // 4*HID
#define KTAG  28
#define START_TAG 26
#define STOP_TAG  27

// ---------------------------------------------------------------------------
// Embedding gather: x[t, 0:256] = word_emb[sentence[t]], x[t, 256:320] = feat_emb[features[t]]
// Output directly in f16 for the WMMA GEMM.
// ---------------------------------------------------------------------------
__global__ void embed_kernel(const int* __restrict__ sent, const int* __restrict__ feat,
                             const float* __restrict__ wemb, const float* __restrict__ femb,
                             _Float16* __restrict__ x) {
  int idx = blockIdx.x * blockDim.x + threadIdx.x;
  if (idx >= T_LEN * 320) return;
  int t = idx / 320, d = idx % 320;
  float v;
  if (d < 256) v = wemb[(size_t)sent[t] * 256 + d];
  else         v = femb[(size_t)feat[t] * 64 + (d - 256)];
  x[idx] = (_Float16)v;
}

// f32 -> f16 flat convert (weights stay [N,K] row-major: each row is a B-fragment column)
__global__ void cvt_f16_kernel(const float* __restrict__ in, _Float16* __restrict__ out, int n) {
  int idx = blockIdx.x * blockDim.x + threadIdx.x;
  if (idx < n) out[idx] = (_Float16)in[idx];
}

// Whh [4H, H] f32 -> WhhT [H, 4H] f16 (k-major so the recurrent matvec is coalesced over gates)
__global__ void tpose_f16_kernel(const float* __restrict__ in, _Float16* __restrict__ out) {
  int idx = blockIdx.x * blockDim.x + threadIdx.x;
  if (idx >= GATES * HID) return;
  int g = idx / HID, k = idx % HID;
  out[(size_t)k * GATES + g] = (_Float16)in[idx];
}

// Concat fwd/bwd hidden states -> f16 input for next layer: x1[t, 0:512]=hf, x1[t,512:1024]=hb
__global__ void concat_kernel(const float* __restrict__ hf, const float* __restrict__ hb,
                              _Float16* __restrict__ x1) {
  int idx = blockIdx.x * blockDim.x + threadIdx.x;
  if (idx >= T_LEN * 1024) return;
  int t = idx / 1024, j = idx % 1024;
  float v = (j < HID) ? hf[(size_t)t * HID + j] : hb[(size_t)t * HID + (j - HID)];
  x1[idx] = (_Float16)v;
}

// ---------------------------------------------------------------------------
// WMMA GEMM: C[M,N] (f32) = A[M,K] (f16 row-major) * W[N,K]^T (f16 row-major).
// Each wave accumulates a 16x64 strip: one A fragment is reused by 4 B tiles
// per k-step (4 v_wmma per 5 fragment-load groups -> ~6.6 MAC/byte vs 4).
// wave32 fragment layouts used here (contiguous k order, same in A and B):
//   A 16x32 : lane L -> row L%16, k=(L/16)*16+[0..15]
//   B 32x16 : lane L -> column L%16 (= row of W, contiguous), k=(L/16)*16+[0..15]
//   C 16x16 : lane L, vgpr v -> (m = v + (L/16)*8, n = L%16)
// Strip id is wave-uniform, so EXEC stays all-1s across v_wmma (ISA requirement).
// ---------------------------------------------------------------------------
__global__ __launch_bounds__(256) void gemm_wmma_kernel(
    const _Float16* __restrict__ A, const _Float16* __restrict__ W,
    float* __restrict__ C, int M, int N, int K) {
  int wave = threadIdx.x >> 5;
  int lane = threadIdx.x & 31;
  int tile = blockIdx.x * 8 + wave;   // 16(M) x 64(N) strips
  int ngrp = N >> 6;
  int mt = tile / ngrp;
  int ng = tile - mt * ngrp;
  if (mt * 16 >= M) return;           // wave-uniform guard
  int lrow = lane & 15;
  int lhi  = lane >> 4;
  const _Float16* arow = A + (size_t)(mt * 16 + lrow) * K + lhi * 16;
  const _Float16* b0 = W + (size_t)(ng * 64 +  0 + lrow) * K + lhi * 16;
  const _Float16* b1 = W + (size_t)(ng * 64 + 16 + lrow) * K + lhi * 16;
  const _Float16* b2 = W + (size_t)(ng * 64 + 32 + lrow) * K + lhi * 16;
  const _Float16* b3 = W + (size_t)(ng * 64 + 48 + lrow) * K + lhi * 16;
  v8f acc0 = {}, acc1 = {}, acc2 = {}, acc3 = {};
  for (int k0 = 0; k0 < K; k0 += 32) {
    union { v16h v; v8h h[2]; } au;
    au.h[0] = *(const v8h*)(arow + k0);
    au.h[1] = *(const v8h*)(arow + k0 + 8);
    v16h vb0 = *(const v16h*)(b0 + k0);
    v16h vb1 = *(const v16h*)(b1 + k0);
    v16h vb2 = *(const v16h*)(b2 + k0);
    v16h vb3 = *(const v16h*)(b3 + k0);
    acc0 = __builtin_amdgcn_wmma_f32_16x16x32_f16(false, au.v, false, vb0, (short)0, acc0, false, false);
    acc1 = __builtin_amdgcn_wmma_f32_16x16x32_f16(false, au.v, false, vb1, (short)0, acc1, false, false);
    acc2 = __builtin_amdgcn_wmma_f32_16x16x32_f16(false, au.v, false, vb2, (short)0, acc2, false, false);
    acc3 = __builtin_amdgcn_wmma_f32_16x16x32_f16(false, au.v, false, vb3, (short)0, acc3, false, false);
  }
  float* crow = C + (size_t)(mt * 16 + lhi * 8) * N + ng * 64 + lrow;
#pragma unroll
  for (int v = 0; v < 8; ++v) {
    crow[(size_t)v * N +  0] = acc0[v];
    crow[(size_t)v * N + 16] = acc1[v];
    crow[(size_t)v * N + 32] = acc2[v];
    crow[(size_t)v * N + 48] = acc3[v];
  }
}

// ---------------------------------------------------------------------------
// Sequential LSTM recurrence. grid.x = 2 (dir 0 = forward, dir 1 = backward).
// 1024 threads; thread tid computes gates 2*tid, 2*tid+1 (h2 loads from WhhT,
// coalesced over gates); threads < 512 then own hidden unit tid (c in register).
// P already holds x @ Wih^T (f32); biases added here. h0 = c0 = 1.0 (torch.ones).
// global_prefetch_b8 hides the next step's P-row fetch behind the matvec.
// ---------------------------------------------------------------------------
__global__ __launch_bounds__(1024) void lstm_rec_kernel(
    const float* __restrict__ Pf, const float* __restrict__ Pb,
    const _Float16* __restrict__ Wtf, const _Float16* __restrict__ Wtb,
    const float* __restrict__ bihf, const float* __restrict__ bhhf,
    const float* __restrict__ bihb, const float* __restrict__ bhhb,
    float* __restrict__ hfout, float* __restrict__ hbout) {
  int dir = blockIdx.x;
  const float*    P   = dir ? Pb   : Pf;
  const _Float16* Wt  = dir ? Wtb  : Wtf;
  const float*    bih = dir ? bihb : bihf;
  const float*    bhh = dir ? bhhb : bhhf;
  float*          ho  = dir ? hbout : hfout;

  __shared__ float sh[HID];
  __shared__ float sg[GATES];

  int tid = threadIdx.x;
  int g0  = tid * 2;
  float bias0 = bih[g0] + bhh[g0];
  float bias1 = bih[g0 + 1] + bhh[g0 + 1];
  float c = 1.0f;
  if (tid < HID) sh[tid] = 1.0f;
  __syncthreads();

  const h2* wp = (const h2*)(Wt + g0);   // Wt[k*GATES + g0], stride GATES/2 h2
  for (int s = 0; s < T_LEN; ++s) {
    int t = dir ? (T_LEN - 1 - s) : s;
    int tn = dir ? (t - 1) : (t + 1);
    if (s + 1 < T_LEN)
      __builtin_prefetch(&P[(size_t)tn * GATES + g0], 0, 3);  // global_prefetch_b8
    float a0 = P[(size_t)t * GATES + g0] + bias0;
    float a1 = P[(size_t)t * GATES + g0 + 1] + bias1;
#pragma unroll 4
    for (int k = 0; k < HID; ++k) {
      float hk = sh[k];
      h2 w = wp[(size_t)k * (GATES / 2)];
      a0 += hk * (float)w.x;
      a1 += hk * (float)w.y;
    }
    sg[g0] = a0;
    sg[g0 + 1] = a1;
    __syncthreads();
    if (tid < HID) {
      float ig = 1.0f / (1.0f + __expf(-sg[tid]));
      float fg = 1.0f / (1.0f + __expf(-sg[HID + tid]));
      float gg = tanhf(sg[2 * HID + tid]);
      float og = 1.0f / (1.0f + __expf(-sg[3 * HID + tid]));
      c = fg * c + ig * gg;
      float h = og * tanhf(c);
      ho[(size_t)t * HID + tid] = h;
      sh[tid] = h;
    }
    __syncthreads();
  }
}

// logits[t,n] = pb[n] + hf[t,:]·pw[n,0:512] + hb[t,:]·pw[n,512:1024]
__global__ void logits_kernel(const float* __restrict__ hf, const float* __restrict__ hb,
                              const float* __restrict__ pw, const float* __restrict__ pb,
                              float* __restrict__ out) {
  int t = blockIdx.x * blockDim.y + threadIdx.y;
  int n = threadIdx.x;
  if (t >= T_LEN || n >= KTAG) return;
  const float* wr  = pw + (size_t)n * 1024;
  const float* hfr = hf + (size_t)t * HID;
  const float* hbr = hb + (size_t)t * HID;
  float acc = pb[n];
  for (int j = 0; j < HID; ++j) acc += hfr[j] * wr[j] + hbr[j] * wr[HID + j];
  out[(size_t)t * KTAG + n] = acc;
}

// Viterbi DP over K=28 tags, single wave. transitions[next, prev].
// argmax tie-break: first index (matches jnp.argmax).
__global__ void viterbi_kernel(const float* __restrict__ feats, const float* __restrict__ trans,
                               float* __restrict__ out, int* __restrict__ bp) {
  __shared__ float fv[32];
  int k = threadIdx.x;
  fv[k] = (k == START_TAG) ? 0.0f : -10000.0f;
  __syncthreads();
  for (int t = 0; t < T_LEN; ++t) {
    float best = -3.0e38f; int bi = 0;
    if (k < KTAG) {
      for (int p = 0; p < KTAG; ++p) {
        float s = fv[p] + trans[k * KTAG + p];
        if (s > best) { best = s; bi = p; }
      }
      best += feats[(size_t)t * KTAG + k];
      bp[(size_t)t * KTAG + k] = bi;
    }
    __syncthreads();
    if (k < KTAG) fv[k] = best;
    __syncthreads();
  }
  if (k == 0) {
    float bs = -3.0e38f; int bt = 0;
    for (int p = 0; p < KTAG; ++p) {
      float s = fv[p] + trans[STOP_TAG * KTAG + p];
      if (s > bs) { bs = s; bt = p; }
    }
    out[0] = bs;
    int tag = bt;
    for (int t = T_LEN - 1; t >= 0; --t) {
      out[1 + t] = (float)tag;          // path[t]
      tag = bp[(size_t)t * KTAG + tag]; // path[t-1] = bps[t][path[t]]
    }
  }
}

// ---------------------------------------------------------------------------
// Host orchestration
// ---------------------------------------------------------------------------
static inline char* carve(char*& p, size_t bytes) {
  char* r = p;
  p += (bytes + 255) & ~(size_t)255;
  return r;
}

extern "C" void kernel_launch(void* const* d_in, const int* in_sizes, int n_in,
                              void* d_out, int out_size, void* d_ws, size_t ws_size,
                              hipStream_t stream) {
  // Input order: sentence, features, word_emb, feat_emb,
  //   layer0: Wih_f, Whh_f, bih_f, bhh_f, Wih_b, Whh_b, bih_b, bhh_b,
  //   layer1: same 8, then proj_w, proj_b, transitions.
  const int*   sent = (const int*)d_in[0];
  const int*   feat = (const int*)d_in[1];
  const float* wemb = (const float*)d_in[2];
  const float* femb = (const float*)d_in[3];
  const float* L0[8]; for (int i = 0; i < 8; ++i) L0[i] = (const float*)d_in[4 + i];
  const float* L1[8]; for (int i = 0; i < 8; ++i) L1[i] = (const float*)d_in[12 + i];
  const float* proj_w = (const float*)d_in[20];
  const float* proj_b = (const float*)d_in[21];
  const float* trans  = (const float*)d_in[22];
  float* out = (float*)d_out;

  // Workspace carve (~103 MB)
  char* p = (char*)d_ws;
  _Float16* x0  = (_Float16*)carve(p, (size_t)T_LEN * 320 * 2);
  _Float16* x1  = (_Float16*)carve(p, (size_t)T_LEN * 1024 * 2);
  _Float16* wihF = (_Float16*)carve(p, (size_t)GATES * 1024 * 2);
  _Float16* wihB = (_Float16*)carve(p, (size_t)GATES * 1024 * 2);
  _Float16* whhF = (_Float16*)carve(p, (size_t)HID * GATES * 2);
  _Float16* whhB = (_Float16*)carve(p, (size_t)HID * GATES * 2);
  float* Pf = (float*)carve(p, (size_t)T_LEN * GATES * 4);
  float* Pb = (float*)carve(p, (size_t)T_LEN * GATES * 4);
  float* hf = (float*)carve(p, (size_t)T_LEN * HID * 4);
  float* hb = (float*)carve(p, (size_t)T_LEN * HID * 4);
  float* lg = (float*)carve(p, (size_t)T_LEN * KTAG * 4);
  int*   bp = (int*)  carve(p, (size_t)T_LEN * KTAG * 4);
  (void)ws_size; (void)in_sizes; (void)n_in; (void)out_size;

  // 1) Embedding gather -> f16
  embed_kernel<<<(T_LEN * 320 + 255) / 256, 256, 0, stream>>>(sent, feat, wemb, femb, x0);

  // ---- Layer 0 (K = 320) ----
  {
    int nW = GATES * 320;
    cvt_f16_kernel<<<(nW + 255) / 256, 256, 0, stream>>>(L0[0], wihF, nW);
    cvt_f16_kernel<<<(nW + 255) / 256, 256, 0, stream>>>(L0[4], wihB, nW);
    tpose_f16_kernel<<<(GATES * HID + 255) / 256, 256, 0, stream>>>(L0[1], whhF);
    tpose_f16_kernel<<<(GATES * HID + 255) / 256, 256, 0, stream>>>(L0[5], whhB);
    int strips = (T_LEN / 16) * (GATES / 64);
    gemm_wmma_kernel<<<(strips + 7) / 8, 256, 0, stream>>>(x0, wihF, Pf, T_LEN, GATES, 320);
    gemm_wmma_kernel<<<(strips + 7) / 8, 256, 0, stream>>>(x0, wihB, Pb, T_LEN, GATES, 320);
    lstm_rec_kernel<<<2, 1024, 0, stream>>>(Pf, Pb, whhF, whhB,
                                            L0[2], L0[3], L0[6], L0[7], hf, hb);
  }

  // concat -> f16 input for layer 1
  concat_kernel<<<(T_LEN * 1024 + 255) / 256, 256, 0, stream>>>(hf, hb, x1);

  // ---- Layer 1 (K = 1024) ----
  {
    int nW = GATES * 1024;
    cvt_f16_kernel<<<(nW + 255) / 256, 256, 0, stream>>>(L1[0], wihF, nW);
    cvt_f16_kernel<<<(nW + 255) / 256, 256, 0, stream>>>(L1[4], wihB, nW);
    tpose_f16_kernel<<<(GATES * HID + 255) / 256, 256, 0, stream>>>(L1[1], whhF);
    tpose_f16_kernel<<<(GATES * HID + 255) / 256, 256, 0, stream>>>(L1[5], whhB);
    int strips = (T_LEN / 16) * (GATES / 64);
    gemm_wmma_kernel<<<(strips + 7) / 8, 256, 0, stream>>>(x1, wihF, Pf, T_LEN, GATES, 1024);
    gemm_wmma_kernel<<<(strips + 7) / 8, 256, 0, stream>>>(x1, wihB, Pb, T_LEN, GATES, 1024);
    lstm_rec_kernel<<<2, 1024, 0, stream>>>(Pf, Pb, whhF, whhB,
                                            L1[2], L1[3], L1[6], L1[7], hf, hb);
  }

  // Projection to tag logits
  dim3 lb(32, 8);
  logits_kernel<<<(T_LEN + 7) / 8, lb, 0, stream>>>(hf, hb, proj_w, proj_b, lg);

  // CRF Viterbi decode (score -> out[0], path -> out[1..T])
  viterbi_kernel<<<1, 32, 0, stream>>>(lg, trans, out, bp);
}
